// MOE_43963285242647
// MI455X (gfx1250) — compile-verified, matrix operand order checked by reading, not run on hardware
//
#include <hip/hip_runtime.h>
#include <math.h>

#define D_DIM 2560
#define H_DIM 10240
#define E_NUM 4
#define NTOK  4096      // B*S
#define BM 128
#define BN 128
#define BK 32
#define BKP 40          // padded LDS row (elements); 80B, multiple of 16B
#define NBUF 3          // async pipeline depth (one-barrier-per-iter safe)

typedef __attribute__((ext_vector_type(16))) __bf16 bf16x16;
typedef __attribute__((ext_vector_type(8)))  __bf16 bf16x8;
typedef __attribute__((ext_vector_type(8)))  float  f32x8;
typedef __attribute__((ext_vector_type(4)))  float  f32x4;
typedef __attribute__((ext_vector_type(4)))  unsigned int u32x4;

__device__ __forceinline__ unsigned short f2bf(float f) {
  unsigned int u = __float_as_uint(f);
  u += 0x7fffu + ((u >> 16) & 1u);          // round-to-nearest-even
  return (unsigned short)(u >> 16);
}
__device__ __forceinline__ float gelu_exact(float x) {
  return 0.5f * x * (1.0f + erff(x * 0.70710678118654752f));
}

// ---- CDNA5 async global->LDS copy (ASYNCcnt path), 16B per lane -----------
__device__ __forceinline__ void async_b128(unsigned lds_off, unsigned long long gaddr) {
  asm volatile("global_load_async_to_lds_b128 %0, %1, off"
               :: "v"(lds_off), "v"(gaddr) : "memory");
}
__device__ __forceinline__ void wait_async_le4() {
#if __has_builtin(__builtin_amdgcn_s_wait_asynccnt)
  __builtin_amdgcn_s_wait_asynccnt(4);
#else
  asm volatile("s_wait_asynccnt 0x4" ::: "memory");
#endif
}
__device__ __forceinline__ void wait_async_le0() {
#if __has_builtin(__builtin_amdgcn_s_wait_asynccnt)
  __builtin_amdgcn_s_wait_asynccnt(0);
#else
  asm volatile("s_wait_asynccnt 0x0" ::: "memory");
#endif
}

// ---------------------------------------------------------------- gating ----
__global__ void moe_gate(const float* __restrict__ X, const float* __restrict__ GW,
                         int* __restrict__ cnt, int* __restrict__ lists,
                         float* __restrict__ sscore) {
  int t = blockIdx.x * 4 + (threadIdx.x >> 5);
  int lane = threadIdx.x & 31;
  const float* xr = X + (size_t)t * D_DIM;
  float g0 = 0.f, g1 = 0.f, g2 = 0.f, g3 = 0.f;
  for (int d = lane; d < D_DIM; d += 32) {
    float xv = xr[d];
    g0 = fmaf(xv, GW[0 * D_DIM + d], g0);
    g1 = fmaf(xv, GW[1 * D_DIM + d], g1);
    g2 = fmaf(xv, GW[2 * D_DIM + d], g2);
    g3 = fmaf(xv, GW[3 * D_DIM + d], g3);
  }
  for (int off = 16; off > 0; off >>= 1) {
    g0 += __shfl_xor(g0, off, 32);
    g1 += __shfl_xor(g1, off, 32);
    g2 += __shfl_xor(g2, off, 32);
    g3 += __shfl_xor(g3, off, 32);
  }
  if (lane == 0) {
    float g[4] = {g0, g1, g2, g3};
    // top_k(-gates, 2): two SMALLEST gates, ties -> lower index first
    int k0 = 0;
    for (int e = 1; e < 4; ++e) if (g[e] < g[k0]) k0 = e;
    int k1 = (k0 == 0) ? 1 : 0;
    for (int e = 0; e < 4; ++e) if (e != k0 && g[e] < g[k1]) k1 = e;
    float a = g[k0], b = g[k1];
    float m = fmaxf(a, b);
    float ea = __expf(a - m), eb = __expf(b - m);
    float inv = 1.f / (ea + eb);
    sscore[t * 2 + 0] = ea * inv;
    sscore[t * 2 + 1] = eb * inv;
    int p0 = atomicAdd(&cnt[k0], 1);
    lists[k0 * NTOK + p0] = t * 2 + 0;
    int p1 = atomicAdd(&cnt[k1], 1);
    lists[k1 * NTOK + p1] = t * 2 + 1;
  }
}

// --------------------------------------------------- prepass: x fp32->bf16 --
__global__ void cvt_x(const float* __restrict__ X, unsigned short* __restrict__ XB) {
  size_t i = ((size_t)blockIdx.x * 256 + threadIdx.x) * 8;
  f32x4 a = *(const f32x4*)(X + i);
  f32x4 b = *(const f32x4*)(X + i + 4);
  u32x4 o;
  o[0] = (unsigned)f2bf(a[0]) | ((unsigned)f2bf(a[1]) << 16);
  o[1] = (unsigned)f2bf(a[2]) | ((unsigned)f2bf(a[3]) << 16);
  o[2] = (unsigned)f2bf(b[0]) | ((unsigned)f2bf(b[1]) << 16);
  o[3] = (unsigned)f2bf(b[2]) | ((unsigned)f2bf(b[3]) << 16);
  *(u32x4*)(XB + i) = o;
}

// ------------------- prepass: weight fp32 [R][C] -> bf16 [C][R] (per expert) -
__global__ void wcvt_t(const float* __restrict__ src, unsigned short* __restrict__ dst,
                       int R, int C) {
  const float* S = src + (size_t)blockIdx.z * R * C;
  unsigned short* Dp = dst + (size_t)blockIdx.z * R * C;
  __shared__ unsigned short t[32][33];
  int r0 = blockIdx.y * 32, c0 = blockIdx.x * 32;
  int tx = threadIdx.x & 31, ty = threadIdx.x >> 5;      // 32 x 8
  for (int rr = ty; rr < 32; rr += 8)
    t[rr][tx] = f2bf(S[(size_t)(r0 + rr) * C + c0 + tx]);
  __syncthreads();
  int kp = threadIdx.x & 15, cc = threadIdx.x >> 4;      // 16 k-pairs x 16 cols, x2
  for (int ci = 0; ci < 2; ++ci) {
    int c = cc + 16 * ci;
    unsigned int v = (unsigned)t[2 * kp][c] | ((unsigned)t[2 * kp + 1][c] << 16);
    *(unsigned int*)(Dp + (size_t)(c0 + c) * R + r0 + 2 * kp) = v;
  }
}

// ------------------------------------------------------------------ fc1 -----
// h[slot,:] = gelu(x[tok] @ w1[e] + b1[e]) bf16; A=XB (gathered), B=W1T [n][k]
__global__ __launch_bounds__(256) void moe_fc1(
    const unsigned short* __restrict__ XB, const unsigned short* __restrict__ W1T,
    const float* __restrict__ B1, const int* __restrict__ cnt,
    const int* __restrict__ lists, unsigned short* __restrict__ HB) {
  constexpr int KD = D_DIM;
  constexpr int KT = KD / BK;                 // 80
  int e = blockIdx.x >> 5;
  int mtile = blockIdx.x & 31;
  int myCnt = cnt[e];
  if (mtile * BM >= myCnt) return;
  int n0 = blockIdx.y * BN;

  __shared__ __align__(16) unsigned short sA[NBUF][BM * BKP];
  __shared__ __align__(16) unsigned short sB[NBUF][BN * BKP];
  __shared__ int sSlot[BM];
  __shared__ int sTok[BM];

  int tid = threadIdx.x;
  if (tid < BM) {
    int p = mtile * BM + tid;
    int pc = (p < myCnt) ? p : (myCnt - 1);
    int slot = lists[e * NTOK + pc];
    sSlot[tid] = slot;
    sTok[tid] = slot >> 1;
  }
  __syncthreads();

  const unsigned short* WT = W1T + (size_t)e * KD * H_DIM;  // [H][D]
  const unsigned TILEB = BM * BKP * 2;                      // 10240 B per buffer
  unsigned sAoff = (unsigned)(uintptr_t)&sA[0][0];
  unsigned sBoff = (unsigned)(uintptr_t)&sB[0][0];

  unsigned long long gA[2], gB[2];
  unsigned lA[2], lB[2];
#pragma unroll
  for (int i = 0; i < 2; ++i) {
    int c = tid + 256 * i;
    int row = c >> 2, sub = c & 3;            // 4 x 16B chunks per 64B row
    gA[i] = (unsigned long long)(uintptr_t)(XB + (size_t)sTok[row] * KD) + sub * 16;
    gB[i] = (unsigned long long)(uintptr_t)(WT + (size_t)(n0 + row) * KD) + sub * 16;
    lA[i] = sAoff + row * (BKP * 2) + sub * 16;
    lB[i] = sBoff + row * (BKP * 2) + sub * 16;
  }
  // prologue: issue kt=0 -> buf0, kt=1 -> buf1
#pragma unroll
  for (int p = 0; p < 2; ++p) {
#pragma unroll
    for (int i = 0; i < 2; ++i) {
      async_b128(lA[i] + p * TILEB, gA[i] + (unsigned long long)p * (BK * 2));
      async_b128(lB[i] + p * TILEB, gB[i] + (unsigned long long)p * (BK * 2));
    }
  }

  int lane = tid & 31, wave = tid >> 5;
  int wm = wave & 1, wn = wave >> 1;          // 2x4 wave grid, wave tile 64x32
  int lr = lane & 15, kh = (lane >> 4) * 8;

  f32x8 acc[4][2];
  for (int i = 0; i < 4; ++i)
    for (int j = 0; j < 2; ++j)
      for (int q = 0; q < 8; ++q) acc[i][j][q] = 0.f;

  for (int kt = 0; kt < KT; ++kt) {
    if (kt + 1 < KT) wait_async_le4(); else wait_async_le0();
    __syncthreads();                           // acquire kt tiles, release kt-1 reads
    if (kt + 2 < KT) {
      int nb = (kt + 2) % NBUF;
      unsigned long long koff = (unsigned long long)(kt + 2) * (BK * 2);
#pragma unroll
      for (int i = 0; i < 2; ++i) {
        async_b128(lA[i] + nb * TILEB, gA[i] + koff);
        async_b128(lB[i] + nb * TILEB, gB[i] + koff);
      }
    }
    int cb = kt % NBUF;
    const unsigned short* a_ = &sA[cb][0];
    const unsigned short* b_ = &sB[cb][0];

    bf16x16 af[4], bfv[2];
#pragma unroll
    for (int fm = 0; fm < 4; ++fm) {
      int row = wm * 64 + fm * 16 + lr;
      bf16x8 lo = *(const bf16x8*)&a_[row * BKP + kh];
      bf16x8 hi = *(const bf16x8*)&a_[row * BKP + 16 + kh];
      af[fm] = __builtin_shufflevector(lo, hi, 0, 1, 2, 3, 4, 5, 6, 7, 8, 9, 10, 11, 12, 13, 14, 15);
    }
#pragma unroll
    for (int fn = 0; fn < 2; ++fn) {
      int col = wn * 32 + fn * 16 + lr;
      bf16x8 lo = *(const bf16x8*)&b_[col * BKP + kh];
      bf16x8 hi = *(const bf16x8*)&b_[col * BKP + 16 + kh];
      bfv[fn] = __builtin_shufflevector(lo, hi, 0, 1, 2, 3, 4, 5, 6, 7, 8, 9, 10, 11, 12, 13, 14, 15);
    }
#pragma unroll
    for (int fm = 0; fm < 4; ++fm)
#pragma unroll
      for (int fn = 0; fn < 2; ++fn)
        acc[fm][fn] = __builtin_amdgcn_wmma_f32_16x16x32_bf16(
            false, af[fm], false, bfv[fn], (short)0, acc[fm][fn], false, false);
  }

#pragma unroll
  for (int fm = 0; fm < 4; ++fm) {
#pragma unroll
    for (int fn = 0; fn < 2; ++fn) {
      int hcol = n0 + wn * 32 + fn * 16 + lr;
      float bias = B1[(size_t)e * H_DIM + hcol];
#pragma unroll
      for (int j = 0; j < 8; ++j) {
        int rowL = wm * 64 + fm * 16 + kh + j;   // M = (lane>=16?8:0)+j
        if (mtile * BM + rowL < myCnt) {
          float hv = gelu_exact(acc[fm][fn][j] + bias);
          HB[(size_t)sSlot[rowL] * H_DIM + hcol] = f2bf(hv);
        }
      }
    }
  }
}

// ------------------------------------------------------------------ fc2 -----
// y[slot,:] = (h[slot] @ w2[e] + b2[e]) * score[slot]; A=HB, B=W2T [n][k]
__global__ __launch_bounds__(256) void moe_fc2(
    const unsigned short* __restrict__ HB, const unsigned short* __restrict__ W2T,
    const float* __restrict__ B2, const int* __restrict__ cnt,
    const int* __restrict__ lists, const float* __restrict__ sscore,
    float* __restrict__ YT) {
  constexpr int KD = H_DIM;
  constexpr int KT = KD / BK;                 // 320
  int e = blockIdx.x >> 5;
  int mtile = blockIdx.x & 31;
  int myCnt = cnt[e];
  if (mtile * BM >= myCnt) return;
  int n0 = blockIdx.y * BN;

  __shared__ __align__(16) unsigned short sA[NBUF][BM * BKP];
  __shared__ __align__(16) unsigned short sB[NBUF][BN * BKP];
  __shared__ int sSlot[BM];
  __shared__ float sSc[BM];

  int tid = threadIdx.x;
  if (tid < BM) {
    int p = mtile * BM + tid;
    int pc = (p < myCnt) ? p : (myCnt - 1);
    int slot = lists[e * NTOK + pc];
    sSlot[tid] = slot;
    sSc[tid] = sscore[slot];
  }
  __syncthreads();

  const unsigned short* WT = W2T + (size_t)e * KD * D_DIM;  // [D][H]
  const unsigned TILEB = BM * BKP * 2;
  unsigned sAoff = (unsigned)(uintptr_t)&sA[0][0];
  unsigned sBoff = (unsigned)(uintptr_t)&sB[0][0];

  unsigned long long gA[2], gB[2];
  unsigned lA[2], lB[2];
#pragma unroll
  for (int i = 0; i < 2; ++i) {
    int c = tid + 256 * i;
    int row = c >> 2, sub = c & 3;
    gA[i] = (unsigned long long)(uintptr_t)(HB + (size_t)sSlot[row] * KD) + sub * 16;
    gB[i] = (unsigned long long)(uintptr_t)(WT + (size_t)(n0 + row) * KD) + sub * 16;
    lA[i] = sAoff + row * (BKP * 2) + sub * 16;
    lB[i] = sBoff + row * (BKP * 2) + sub * 16;
  }
#pragma unroll
  for (int p = 0; p < 2; ++p) {
#pragma unroll
    for (int i = 0; i < 2; ++i) {
      async_b128(lA[i] + p * TILEB, gA[i] + (unsigned long long)p * (BK * 2));
      async_b128(lB[i] + p * TILEB, gB[i] + (unsigned long long)p * (BK * 2));
    }
  }

  int lane = tid & 31, wave = tid >> 5;
  int wm = wave & 1, wn = wave >> 1;
  int lr = lane & 15, kh = (lane >> 4) * 8;

  f32x8 acc[4][2];
  for (int i = 0; i < 4; ++i)
    for (int j = 0; j < 2; ++j)
      for (int q = 0; q < 8; ++q) acc[i][j][q] = 0.f;

  for (int kt = 0; kt < KT; ++kt) {
    if (kt + 1 < KT) wait_async_le4(); else wait_async_le0();
    __syncthreads();
    if (kt + 2 < KT) {
      int nb = (kt + 2) % NBUF;
      unsigned long long koff = (unsigned long long)(kt + 2) * (BK * 2);
#pragma unroll
      for (int i = 0; i < 2; ++i) {
        async_b128(lA[i] + nb * TILEB, gA[i] + koff);
        async_b128(lB[i] + nb * TILEB, gB[i] + koff);
      }
    }
    int cb = kt % NBUF;
    const unsigned short* a_ = &sA[cb][0];
    const unsigned short* b_ = &sB[cb][0];

    bf16x16 af[4], bfv[2];
#pragma unroll
    for (int fm = 0; fm < 4; ++fm) {
      int row = wm * 64 + fm * 16 + lr;
      bf16x8 lo = *(const bf16x8*)&a_[row * BKP + kh];
      bf16x8 hi = *(const bf16x8*)&a_[row * BKP + 16 + kh];
      af[fm] = __builtin_shufflevector(lo, hi, 0, 1, 2, 3, 4, 5, 6, 7, 8, 9, 10, 11, 12, 13, 14, 15);
    }
#pragma unroll
    for (int fn = 0; fn < 2; ++fn) {
      int col = wn * 32 + fn * 16 + lr;
      bf16x8 lo = *(const bf16x8*)&b_[col * BKP + kh];
      bf16x8 hi = *(const bf16x8*)&b_[col * BKP + 16 + kh];
      bfv[fn] = __builtin_shufflevector(lo, hi, 0, 1, 2, 3, 4, 5, 6, 7, 8, 9, 10, 11, 12, 13, 14, 15);
    }
#pragma unroll
    for (int fm = 0; fm < 4; ++fm)
#pragma unroll
      for (int fn = 0; fn < 2; ++fn)
        acc[fm][fn] = __builtin_amdgcn_wmma_f32_16x16x32_bf16(
            false, af[fm], false, bfv[fn], (short)0, acc[fm][fn], false, false);
  }

#pragma unroll
  for (int fm = 0; fm < 4; ++fm) {
#pragma unroll
    for (int fn = 0; fn < 2; ++fn) {
      int col = n0 + wn * 32 + fn * 16 + lr;
      float bias = B2[(size_t)e * D_DIM + col];
#pragma unroll
      for (int j = 0; j < 8; ++j) {
        int rowL = wm * 64 + fm * 16 + kh + j;
        if (mtile * BM + rowL < myCnt) {
          float yv = (acc[fm][fn][j] + bias) * sSc[rowL];
          YT[(size_t)sSlot[rowL] * D_DIM + col] = yv;
        }
      }
    }
  }
}

// -------------------------------------------------------------- combine -----
__global__ void moe_combine(const float* __restrict__ YT, float* __restrict__ OUT) {
  size_t i = ((size_t)blockIdx.x * 256 + threadIdx.x) * 4;   // i < NTOK*D_DIM
  size_t t = i / D_DIM;
  f32x4 a = *(const f32x4*)(YT + i + t * D_DIM);
  f32x4 b = *(const f32x4*)(YT + i + t * D_DIM + D_DIM);
  f32x4 o;
  o[0] = a[0] + b[0]; o[1] = a[1] + b[1]; o[2] = a[2] + b[2]; o[3] = a[3] + b[3];
  *(f32x4*)(OUT + i) = o;
}

// --------------------------------------------------------------- launch -----
extern "C" void kernel_launch(void* const* d_in, const int* in_sizes, int n_in,
                              void* d_out, int out_size, void* d_ws, size_t ws_size,
                              hipStream_t stream) {
  (void)in_sizes; (void)n_in; (void)out_size; (void)ws_size;
  const float* x      = (const float*)d_in[0];
  const float* gate_w = (const float*)d_in[1];
  const float* w1     = (const float*)d_in[2];
  const float* b1     = (const float*)d_in[3];
  const float* w2     = (const float*)d_in[4];
  const float* b2     = (const float*)d_in[5];
  float* out = (float*)d_out;

  char* ws = (char*)d_ws;
  size_t off = 0;
  int* cnt = (int*)(ws + off);            off += 256;
  int* lists = (int*)(ws + off);          off += (size_t)E_NUM * NTOK * 4;
  float* sscore = (float*)(ws + off);     off += (size_t)NTOK * 2 * 4;
  off = (off + 255) & ~(size_t)255;
  unsigned short* xb  = (unsigned short*)(ws + off); off += (size_t)NTOK * D_DIM * 2;
  unsigned short* w1t = (unsigned short*)(ws + off); off += (size_t)E_NUM * D_DIM * H_DIM * 2;
  unsigned short* w2t = (unsigned short*)(ws + off); off += (size_t)E_NUM * D_DIM * H_DIM * 2;
  unsigned short* hb  = (unsigned short*)(ws + off); off += (size_t)NTOK * 2 * H_DIM * 2;
  float* yt = (float*)(ws + off);         off += (size_t)NTOK * 2 * D_DIM * 4;

  hipMemsetAsync(cnt, 0, E_NUM * sizeof(int), stream);
  moe_gate<<<NTOK / 4, 128, 0, stream>>>(x, gate_w, cnt, lists, sscore);
  cvt_x<<<(unsigned)(((size_t)NTOK * D_DIM) / (256 * 8)), 256, 0, stream>>>(x, xb);
  wcvt_t<<<dim3(H_DIM / 32, D_DIM / 32, E_NUM), 256, 0, stream>>>(w1, w1t, D_DIM, H_DIM);
  wcvt_t<<<dim3(D_DIM / 32, H_DIM / 32, E_NUM), 256, 0, stream>>>(w2, w2t, H_DIM, D_DIM);
  moe_fc1<<<dim3(E_NUM * 32, H_DIM / BN), 256, 0, stream>>>(xb, w1t, b1, cnt, lists, hb);
  moe_fc2<<<dim3(E_NUM * 32, D_DIM / BN), 256, 0, stream>>>(hb, w2t, b2, cnt, lists, sscore, yt);
  moe_combine<<<(unsigned)(((size_t)NTOK * D_DIM) / (256 * 4)), 256, 0, stream>>>(yt, out);
}
